// CausalSelfAttention_7799660609849
// MI455X (gfx1250) — matrix operand order repaired and checked by reading, not compiled, not run on hardware
//
#include <hip/hip_runtime.h>

// ---------------- problem constants ----------------
#define Bb   4
#define Tt   2048
#define Cc   1024
#define Hh   16
#define Dd   64
#define Mm   (Bb * Tt)        // 8192
#define N3C  (3 * Cc)         // 3072

typedef _Float16 h16;
typedef __attribute__((ext_vector_type(16))) _Float16 v16h;
typedef __attribute__((ext_vector_type(8)))  _Float16 v8h;
typedef __attribute__((ext_vector_type(8)))  float    v8f;
typedef __attribute__((ext_vector_type(4)))  int      v4i;

#if defined(__has_builtin)
#if __has_builtin(__builtin_amdgcn_global_load_async_to_lds_b128)
#define HAVE_ASYNC_LDS 1
#endif
#endif

union FragU { v16h v; v8h h[2]; };

// A-matrix 16x32 f16 fragment (ISA 7.12.2): lanes 0-15 row M=lane, K 0..7 (v0..3)
// and K 16..23 (v4..7); lanes 16-31 same rows, K 8..15 / 24..31.
__device__ __forceinline__ v16h load_frag_a(const h16* p, int ld, int r0, int c0) {
    const int lane = threadIdx.x & 31;
    const int hs   = lane >> 4;
    const int l    = lane & 15;
    const h16* base = p + (size_t)(r0 + l) * ld + c0 + hs * 8;
    FragU u;
    u.h[0] = *(const v8h*)(base);
    u.h[1] = *(const v8h*)(base + 16);
    return u.v;
}

// B-matrix 32x16 f16 fragment loaded from B^T stored row-major [N, K]:
// lanes 0-15: N=n0+lane, K k0+0..7 and k0+16..23; lanes 16-31: K k0+8..15 and k0+24..31.
__device__ __forceinline__ v16h load_frag_b(const h16* p, int ld, int n0, int k0) {
    const int lane = threadIdx.x & 31;
    const int hs   = lane >> 4;
    const int l    = lane & 15;
    const h16* base = p + (size_t)(n0 + l) * ld + k0 + hs * 8;
    FragU u;
    u.h[0] = *(const v8h*)(base);
    u.h[1] = *(const v8h*)(base + 16);
    return u.v;
}

__device__ __forceinline__ v8f wmma16(v16h a, v16h b, v8f c) {
    return __builtin_amdgcn_wmma_f32_16x16x32_f16(false, a, false, b,
                                                  (short)0, c, false, false);
}

// 16B cooperative copy global -> LDS (async on CDNA5, ds-store fallback otherwise)
__device__ __forceinline__ void stage16(const h16* gsrc, h16* ldst) {
#if HAVE_ASYNC_LDS
    // builtin signature (from hipcc diagnostic): param0 = v4i in AS1, param1 = v4i in AS3
    typedef v4i __attribute__((address_space(1)))* gp_t;
    typedef v4i __attribute__((address_space(3)))* lp_t;
    __builtin_amdgcn_global_load_async_to_lds_b128((gp_t)gsrc, (lp_t)ldst, 0, 0);
#else
    *(v8h*)ldst = *(const v8h*)gsrc;
#endif
}

__device__ __forceinline__ void stage_wait() {
#if HAVE_ASYNC_LDS
#if __has_builtin(__builtin_amdgcn_s_wait_asynccnt)
    __builtin_amdgcn_s_wait_asynccnt(0);
#else
    asm volatile("s_wait_asynccnt 0" ::: "memory");
#endif
#endif
}

// ---------------- stage 0: conversions ----------------
__global__ void k_f32_to_f16(const float* __restrict__ in, h16* __restrict__ out, int n) {
    int i = blockIdx.x * blockDim.x + threadIdx.x;
    if (i < n) out[i] = (h16)in[i];
}

// in: [K, N] f32 row-major  ->  out: [N, K] f16 row-major
__global__ void k_transpose_f16(const float* __restrict__ in, h16* __restrict__ out,
                                int K, int N) {
    int i = blockIdx.x * blockDim.x + threadIdx.x;
    if (i < K * N) {
        int k = i / N, n = i - k * N;
        out[(size_t)n * K + k] = (h16)in[i];
    }
}

// ---------------- stage 1: QKV GEMM ----------------
// 256 threads = 8 waves (2M x 4N), wave tile 64x64 => 16 WMMA / 16 b128 loads per k-step
__global__ __launch_bounds__(256) void k_gemm_qkv(
    const h16* __restrict__ xh, const h16* __restrict__ wT,
    const float* __restrict__ bias,
    h16* __restrict__ Qh, h16* __restrict__ Kh, h16* __restrict__ Vt) {
    const int lane = threadIdx.x & 31;
    const int w    = threadIdx.x >> 5;
    const int wm   = w >> 2, wn = w & 3;
    const int m0 = blockIdx.y * 128 + wm * 64;
    const int n0 = blockIdx.x * 256 + wn * 64;

    v8f acc[4][4];
#pragma unroll
    for (int i = 0; i < 4; ++i)
#pragma unroll
        for (int j = 0; j < 4; ++j) acc[i][j] = v8f{};

    for (int k0 = 0; k0 < Cc; k0 += 32) {
        v16h fa[4], fb[4];
#pragma unroll
        for (int i = 0; i < 4; ++i) fa[i] = load_frag_a(xh, Cc, m0 + 16 * i, k0);
#pragma unroll
        for (int j = 0; j < 4; ++j) fb[j] = load_frag_b(wT, Cc, n0 + 16 * j, k0);
        if (k0 + 32 < Cc) {   // hint next k-tile into cache (global_prefetch_b8)
            __builtin_prefetch(xh + (size_t)(m0 + 2 * lane) * Cc + k0 + 32, 0, 1);
            __builtin_prefetch(wT + (size_t)(n0 + 2 * lane) * Cc + k0 + 32, 0, 1);
        }
#pragma unroll
        for (int i = 0; i < 4; ++i)
#pragma unroll
            for (int j = 0; j < 4; ++j) acc[i][j] = wmma16(fa[i], fb[j], acc[i][j]);
    }

    const int hi8 = (lane >> 4) * 8;
    const int nl  = lane & 15;
#pragma unroll
    for (int i = 0; i < 4; ++i)
#pragma unroll
        for (int jn = 0; jn < 4; ++jn)
#pragma unroll
            for (int j = 0; j < 8; ++j) {
                int m = m0 + 16 * i + j + hi8;
                int n = n0 + 16 * jn + nl;
                float val = acc[i][jn][j] + bias[n];
                int b = m >> 11, t = m & 2047;          // /T, %T
                int which = n >> 10, c = n & 1023;      // /C, %C
                int hh = c >> 6, d = c & 63;            // /D, %D
                h16 hv = (h16)val;
                size_t bh = (size_t)(b * Hh + hh);
                if (which == 0)      Qh[(bh * Tt + t) * Dd + d] = hv;
                else if (which == 1) Kh[(bh * Tt + t) * Dd + d] = hv;
                else                 Vt[(bh * Dd + d) * Tt + t] = hv;
            }
}

// ---------------- stage 2: flash attention ----------------
// WG = 128 q rows (8 waves x 16). Key blocks strictly below the diagonal block-row
// are needed unmasked by ALL waves -> cooperative async staging of K/V into LDS.
// The <=4 diagonal blocks per wave use private global loads + causal mask.
__global__ __launch_bounds__(256) void k_attn(
    const h16* __restrict__ Qh, const h16* __restrict__ Kh,
    const h16* __restrict__ Vt, h16* __restrict__ yh) {
    __shared__ __align__(16) h16 Ks[32 * 64];        // K rows [32 k][64 d]
    __shared__ __align__(16) h16 Vs[64 * 32];        // V^T rows [64 d][32 k]
    __shared__ __align__(16) h16 Plds[8][16 * 32];   // per-wave P staging

    const int tid  = threadIdx.x;
    const int lane = tid & 31;
    const int w    = tid >> 5;
    const int bh   = blockIdx.y;            // b*H + h
    const int b    = bh >> 4, h = bh & 15;
    const int qblk = blockIdx.x;
    const int q0   = qblk * 128 + w * 16;

    const h16* Qp = Qh + (size_t)bh * Tt * Dd;
    const h16* Kp = Kh + (size_t)bh * Tt * Dd;
    const h16* Vp = Vt + (size_t)bh * Dd * Tt;

    const v16h aQ0 = load_frag_a(Qp, Dd, q0, 0);
    const v16h aQ1 = load_frag_a(Qp, Dd, q0, 32);

    const int hi8 = (lane >> 4) * 8;
    const int nl  = lane & 15;

    float m_r[8], l_r[8];
    v8f acc[4];
#pragma unroll
    for (int j = 0; j < 8; ++j) { m_r[j] = -3.0e38f; l_r[j] = 0.0f; }
#pragma unroll
    for (int d = 0; d < 4; ++d) acc[d] = v8f{};

    // staging coordinates (one 16B chunk per thread per tile)
    const int krow = tid >> 3, kcol = (tid & 7) * 8;   // K: 32x64
    const int vrow = tid >> 2, vcol = (tid & 3) * 8;   // V: 64x32

    // ---- phase 1: unmasked blocks, cooperative LDS staging ----
    for (int kb = 0; kb < qblk * 128; kb += 32) {
        stage16(Kp + (size_t)(kb + krow) * Dd + kcol, &Ks[krow * 64 + kcol]);
        stage16(Vp + (size_t)vrow * Tt + kb + vcol,   &Vs[vrow * 32 + vcol]);
        stage_wait();
        __syncthreads();

        v8f s[2];
#pragma unroll
        for (int t2 = 0; t2 < 2; ++t2) {
            v16h bK0 = load_frag_b(Ks, Dd, t2 * 16, 0);
            v16h bK1 = load_frag_b(Ks, Dd, t2 * 16, 32);
            v8f sv = v8f{};
            sv = wmma16(aQ0, bK0, sv);
            sv = wmma16(aQ1, bK1, sv);
#pragma unroll
            for (int j = 0; j < 8; ++j) sv[j] *= 0.125f;   // 1/sqrt(64), no mask
            s[t2] = sv;
        }

        float mnew[8], sc[8];
#pragma unroll
        for (int j = 0; j < 8; ++j) {
            float tm = fmaxf(s[0][j], s[1][j]);
#pragma unroll
            for (int msk = 1; msk <= 8; msk <<= 1)
                tm = fmaxf(tm, __shfl_xor(tm, msk, 32));
            mnew[j] = fmaxf(m_r[j], tm);
            sc[j]   = __expf(m_r[j] - mnew[j]);
            l_r[j] *= sc[j];
            m_r[j]  = mnew[j];
        }
#pragma unroll
        for (int d = 0; d < 4; ++d)
#pragma unroll
            for (int j = 0; j < 8; ++j) acc[d][j] *= sc[j];

#pragma unroll
        for (int j = 0; j < 8; ++j) {
            float p0 = __expf(s[0][j] - mnew[j]);
            float p1 = __expf(s[1][j] - mnew[j]);
            int r = j + hi8;
            Plds[w][r * 32 + nl]      = (h16)p0;
            Plds[w][r * 32 + 16 + nl] = (h16)p1;
            float ps = p0 + p1;
#pragma unroll
            for (int msk = 1; msk <= 8; msk <<= 1)
                ps += __shfl_xor(ps, msk, 32);
            l_r[j] += ps;
        }

        v16h aP = load_frag_a(&Plds[w][0], 32, 0, 0);
#pragma unroll
        for (int dt = 0; dt < 4; ++dt) {
            v16h bV = load_frag_b(Vs, 32, dt * 16, 0);
            acc[dt] = wmma16(aP, bV, acc[dt]);
        }
        __syncthreads();   // protect Ks/Vs before next staging
    }

    // ---- phase 2: diagonal blocks, private loads + causal mask ----
    for (int kb = qblk * 128; kb < q0 + 16; kb += 32) {
        v8f s[2];
#pragma unroll
        for (int t2 = 0; t2 < 2; ++t2) {
            const int ks = kb + t2 * 16;
            v16h bK0 = load_frag_b(Kp, Dd, ks, 0);
            v16h bK1 = load_frag_b(Kp, Dd, ks, 32);
            v8f sv = v8f{};
            sv = wmma16(aQ0, bK0, sv);
            sv = wmma16(aQ1, bK1, sv);
#pragma unroll
            for (int j = 0; j < 8; ++j) {
                int col = ks + nl;
                int row = q0 + j + hi8;
                float x = sv[j] * 0.125f;
                if (col > row) x += -1.0e10f;   // additive causal mask
                sv[j] = x;
            }
            s[t2] = sv;
        }

        float mnew[8], sc[8];
#pragma unroll
        for (int j = 0; j < 8; ++j) {
            float tm = fmaxf(s[0][j], s[1][j]);
#pragma unroll
            for (int msk = 1; msk <= 8; msk <<= 1)
                tm = fmaxf(tm, __shfl_xor(tm, msk, 32));
            mnew[j] = fmaxf(m_r[j], tm);
            sc[j]   = __expf(m_r[j] - mnew[j]);
            l_r[j] *= sc[j];
            m_r[j]  = mnew[j];
        }
#pragma unroll
        for (int d = 0; d < 4; ++d)
#pragma unroll
            for (int j = 0; j < 8; ++j) acc[d][j] *= sc[j];

#pragma unroll
        for (int j = 0; j < 8; ++j) {
            float p0 = __expf(s[0][j] - mnew[j]);
            float p1 = __expf(s[1][j] - mnew[j]);
            int r = j + hi8;
            Plds[w][r * 32 + nl]      = (h16)p0;
            Plds[w][r * 32 + 16 + nl] = (h16)p1;
            float ps = p0 + p1;
#pragma unroll
            for (int msk = 1; msk <= 8; msk <<= 1)
                ps += __shfl_xor(ps, msk, 32);
            l_r[j] += ps;
        }

        v16h aP = load_frag_a(&Plds[w][0], 32, 0, 0);
#pragma unroll
        for (int dt = 0; dt < 4; ++dt) {
            v16h bV = load_frag_b(Vp, Tt, dt * 16, kb);
            acc[dt] = wmma16(aP, bV, acc[dt]);
        }
    }

    float inv[8];
#pragma unroll
    for (int j = 0; j < 8; ++j) inv[j] = 1.0f / l_r[j];
#pragma unroll
    for (int dt = 0; dt < 4; ++dt)
#pragma unroll
        for (int j = 0; j < 8; ++j) {
            int row = q0 + j + hi8;
            int col = h * Dd + dt * 16 + nl;
            yh[((size_t)b * Tt + row) * Cc + col] = (h16)(acc[dt][j] * inv[j]);
        }
}

// ---------------- stage 3: output projection ----------------
__global__ __launch_bounds__(256) void k_gemm_proj(
    const h16* __restrict__ yh, const h16* __restrict__ wpT,
    const float* __restrict__ bias, float* __restrict__ out) {
    const int lane = threadIdx.x & 31;
    const int w    = threadIdx.x >> 5;
    const int wm   = w >> 2, wn = w & 3;
    const int m0 = blockIdx.y * 128 + wm * 64;
    const int n0 = blockIdx.x * 256 + wn * 64;

    v8f acc[4][4];
#pragma unroll
    for (int i = 0; i < 4; ++i)
#pragma unroll
        for (int j = 0; j < 4; ++j) acc[i][j] = v8f{};

    for (int k0 = 0; k0 < Cc; k0 += 32) {
        v16h fa[4], fb[4];
#pragma unroll
        for (int i = 0; i < 4; ++i) fa[i] = load_frag_a(yh, Cc, m0 + 16 * i, k0);
#pragma unroll
        for (int j = 0; j < 4; ++j) fb[j] = load_frag_b(wpT, Cc, n0 + 16 * j, k0);
        if (k0 + 32 < Cc) {
            __builtin_prefetch(yh  + (size_t)(m0 + 2 * lane) * Cc + k0 + 32, 0, 1);
            __builtin_prefetch(wpT + (size_t)(n0 + 2 * lane) * Cc + k0 + 32, 0, 1);
        }
#pragma unroll
        for (int i = 0; i < 4; ++i)
#pragma unroll
            for (int j = 0; j < 4; ++j) acc[i][j] = wmma16(fa[i], fb[j], acc[i][j]);
    }

    const int hi8 = (lane >> 4) * 8;
    const int nl  = lane & 15;
#pragma unroll
    for (int i = 0; i < 4; ++i)
#pragma unroll
        for (int jn = 0; jn < 4; ++jn)
#pragma unroll
            for (int j = 0; j < 8; ++j) {
                int m = m0 + 16 * i + j + hi8;
                int n = n0 + 16 * jn + nl;
                out[(size_t)m * Cc + n] = acc[i][jn][j] + bias[n];
            }
}

// ---------------- workspace layout (bytes) ----------------
static constexpr size_t OFF_XH  = 0;                                   // 8192x1024 f16
static constexpr size_t OFF_WT  = OFF_XH  + (size_t)Mm * Cc * 2;       // 3072x1024 f16
static constexpr size_t OFF_WPT = OFF_WT  + (size_t)N3C * Cc * 2;      // 1024x1024 f16
static constexpr size_t OFF_Q   = OFF_WPT + (size_t)Cc * Cc * 2;
static constexpr size_t OFF_K   = OFF_Q   + (size_t)Mm * Cc * 2;
static constexpr size_t OFF_V   = OFF_K   + (size_t)Mm * Cc * 2;
static constexpr size_t OFF_Y   = OFF_V   + (size_t)Mm * Cc * 2;

extern "C" void kernel_launch(void* const* d_in, const int* in_sizes, int n_in,
                              void* d_out, int out_size, void* d_ws, size_t ws_size,
                              hipStream_t stream) {
    (void)in_sizes; (void)n_in; (void)out_size; (void)ws_size;
    const float* x      = (const float*)d_in[0];
    const float* w_attn = (const float*)d_in[1];
    const float* b_attn = (const float*)d_in[2];
    const float* w_proj = (const float*)d_in[3];
    const float* b_proj = (const float*)d_in[4];
    float* out = (float*)d_out;

    char* ws = (char*)d_ws;
    h16* xh  = (h16*)(ws + OFF_XH);
    h16* wT  = (h16*)(ws + OFF_WT);
    h16* wpT = (h16*)(ws + OFF_WPT);
    h16* Qh  = (h16*)(ws + OFF_Q);
    h16* Kh  = (h16*)(ws + OFF_K);
    h16* Vt  = (h16*)(ws + OFF_V);
    h16* yh  = (h16*)(ws + OFF_Y);

    {
        int n = Mm * Cc;
        k_f32_to_f16<<<(n + 255) / 256, 256, 0, stream>>>(x, xh, n);
    }
    {
        int n = Cc * N3C;
        k_transpose_f16<<<(n + 255) / 256, 256, 0, stream>>>(w_attn, wT, Cc, N3C);
    }
    {
        int n = Cc * Cc;
        k_transpose_f16<<<(n + 255) / 256, 256, 0, stream>>>(w_proj, wpT, Cc, Cc);
    }

    // QKV GEMM: 8192 x 3072, K=1024 (block tile 128x256)
    k_gemm_qkv<<<dim3(N3C / 256, Mm / 128), 256, 0, stream>>>(xh, wT, b_attn, Qh, Kh, Vt);

    // flash attention: one WG per (b,h, 128 query rows)
    k_attn<<<dim3(Tt / 128, Bb * Hh), 256, 0, stream>>>(Qh, Kh, Vt, yh);

    // projection GEMM: 8192 x 1024, K=1024 -> fp32 out
    k_gemm_proj<<<dim3(Cc / 256, Mm / 128), 256, 0, stream>>>(yh, wpT, b_proj, out);
}
